// LRGAttention_188978561334
// MI455X (gfx1250) — matrix operand order repaired and checked by reading, not compiled, not run on hardware
//
#include <hip/hip_runtime.h>

// ---------------- types ----------------
typedef __attribute__((ext_vector_type(16))) _Float16 v16h;
typedef __attribute__((ext_vector_type(8)))  _Float16 h8v;
typedef __attribute__((ext_vector_type(4)))  _Float16 h4v;
typedef __attribute__((ext_vector_type(8)))  float    v8f;

// ---------------- problem constants ----------------
#define BATCH   8
#define NLOC    8192
#define NREG    512
#define NTOT    8705        // 8192 + 512 + 1
#define NPAD    9216        // NTOT padded to 18*512 (kv split-K chunks)
#define CDIM    256
#define C3      768
#define HEADS   4
#define HD      64
#define QSCALE  0.125f      // HD^-0.5

// ---------------- WMMA fragment loaders (gfx1250 wave32 layouts) ----------
// A fragment 16x32 f16: lane(0..15)=row M, lanes 16..31 repeat rows with the
// other K half: lane<16 holds K {k0..k0+7, k0+16..k0+23}; lane>=16 holds
// {k0+8..k0+15, k0+24..k0+31}. Source must be [M][K] row-major (ld halfs).
static __device__ __forceinline__ v16h
load_afrag(const _Float16* __restrict__ base, int ld, int row0, int k0, int lane) {
    int l16 = lane & 15, hi = (lane >> 4) & 1;
    const _Float16* p = base + (size_t)(row0 + l16) * (size_t)ld + k0 + hi * 8;
    union { v16h v; h8v h[2]; } u;
    u.h[0] = *(const h8v*)(p);
    u.h[1] = *(const h8v*)(p + 16);
    return u.v;
}

// B fragment 32x16 f16: lane(0..15)=col N holding K {k0..k0+7, k0+16..k0+23},
// lanes 16..31 hold K {k0+8..k0+15, k0+24..k0+31}. Source must be stored [N][K] (transposed),
// ld halfs per column.
static __device__ __forceinline__ v16h
load_bfrag(const _Float16* __restrict__ base, int ld, int col0, int k0, int lane) {
    int l16 = lane & 15, hi = (lane >> 4) & 1;
    const _Float16* p = base + (size_t)(col0 + l16) * (size_t)ld + k0 + hi * 8;
    union { v16h v; h8v h[2]; } u;
    u.h[0] = *(const h8v*)(p);
    u.h[1] = *(const h8v*)(p + 16);
    return u.v;
}

static __device__ __forceinline__ v8f
wmma_f16(v16h a, v16h b, v8f c) {
    return __builtin_amdgcn_wmma_f32_16x16x32_f16(false, a, false, b, (short)0, c,
                                                  false, false);
}

// ---------------- prep kernels ----------------
// Concatenate local|region|global tokens into f16 Xh[b][n][256] (row-major).
__global__ void conv_tokens_kernel(const float* __restrict__ loc,
                                   const float* __restrict__ reg,
                                   const float* __restrict__ glob,
                                   _Float16* __restrict__ Xh) {
    size_t i = (size_t)blockIdx.x * blockDim.x + threadIdx.x;
    size_t total4 = (size_t)BATCH * NTOT * CDIM / 4;
    if (i >= total4) return;
    size_t flat = i * 4;
    int b = (int)(flat / ((size_t)NTOT * CDIM));
    size_t rem = flat % ((size_t)NTOT * CDIM);
    int n = (int)(rem / CDIM);
    int c = (int)(rem % CDIM);
    const float* src;
    if (n < NLOC)            src = loc  + (((size_t)b * NLOC + n) * CDIM + c);
    else if (n < NLOC + NREG) src = reg + (((size_t)b * NREG + (n - NLOC)) * CDIM + c);
    else                     src = glob + ((size_t)b * CDIM + c);
    float4 v = *(const float4*)src;
    h4v o;
    o[0] = (_Float16)v.x; o[1] = (_Float16)v.y;
    o[2] = (_Float16)v.z; o[3] = (_Float16)v.w;
    *(h4v*)(Xh + ((size_t)b * NPAD + n) * CDIM + c) = o;
}

// Transpose+convert QKV weights: WhT[s][col][k] = (f16) W_s[k][col].
__global__ void conv_w_kernel(const float* __restrict__ Wl,
                              const float* __restrict__ Wr,
                              const float* __restrict__ Wg,
                              _Float16* __restrict__ WhT) {
    size_t i = (size_t)blockIdx.x * blockDim.x + threadIdx.x;
    if (i >= (size_t)3 * C3 * CDIM) return;
    int s   = (int)(i / ((size_t)C3 * CDIM));
    size_t r = i % ((size_t)C3 * CDIM);
    int col = (int)(r / CDIM);
    int k   = (int)(r % CDIM);
    const float* W = (s == 0) ? Wl : ((s == 1) ? Wr : Wg);
    WhT[i] = (_Float16)W[(size_t)k * C3 + col];
}

// Zero the padded tail columns [8704, NPAD) of kT and vT so the split-K
// reduction over padded chunks contributes zero.
__global__ void zero_tail_kernel(_Float16* __restrict__ kT,
                                 _Float16* __restrict__ vT) {
    size_t i = (size_t)blockIdx.x * blockDim.x + threadIdx.x;
    size_t total8 = (size_t)BATCH * CDIM * (NPAD - 8704) / 8;
    if (i >= total8) return;
    size_t base = i * 8;
    size_t row = base / (NPAD - 8704);
    size_t off = base % (NPAD - 8704);
    size_t addr = row * NPAD + 8704 + off;
    h8v z = {};
    *(h8v*)(kT + addr) = z;
    *(h8v*)(vT + addr) = z;
}

__global__ void zero_kv_kernel(float* __restrict__ kv) {
    size_t i = (size_t)blockIdx.x * blockDim.x + threadIdx.x;
    if (i >= (size_t)BATCH * HEADS * HD * HD / 4) return;
    float4 z; z.x = z.y = z.z = z.w = 0.f;
    ((float4*)kv)[i] = z;
}

// ---------------- Kernel A: QKV projection (f16 WMMA, f32 accum) ----------
// grid (12 colblk, 69 rowblk, 8 batch), 256 threads = 8 waves.
// Workgroup tile: 128 rows x 64 cols; wave tile 32x32 (2x2 WMMA accums).
// Writes: q (scaled, row-major qh[b][n][c]), kT[b][d][n], vT[b][e][n].
__global__ void __launch_bounds__(256)
qkv_gemm_kernel(const _Float16* __restrict__ Xh,
                const _Float16* __restrict__ WhT,
                const float* __restrict__ bl,
                const float* __restrict__ br,
                const float* __restrict__ bg,
                _Float16* __restrict__ qh,
                _Float16* __restrict__ kT,
                _Float16* __restrict__ vT) {
    int cblk = blockIdx.x, nblk = blockIdx.y, b = blockIdx.z;
    int seg = (nblk < 64) ? 0 : ((nblk < 68) ? 1 : 2);
    const _Float16* Wt = WhT + (size_t)seg * C3 * CDIM;     // [col][k]
    const float* bias = (seg == 0) ? bl : ((seg == 1) ? br : bg);

    int tid = threadIdx.x, wave = tid >> 5, lane = tid & 31;
    int wr = wave & 3, wc = wave >> 2;
    int row_base = nblk * 128 + wr * 32;                    // token row in batch
    int col_base = cblk * 64 + wc * 32;                     // output column
    const _Float16* Xrow = Xh + (size_t)b * NPAD * CDIM;

    v8f acc[2][2] = {};
#pragma unroll
    for (int k0 = 0; k0 < CDIM; k0 += 32) {
        v16h a0 = load_afrag(Xrow, CDIM, row_base,      k0, lane);
        v16h a1 = load_afrag(Xrow, CDIM, row_base + 16, k0, lane);
        v16h b0 = load_bfrag(Wt,   CDIM, col_base,      k0, lane);
        v16h b1 = load_bfrag(Wt,   CDIM, col_base + 16, k0, lane);
        acc[0][0] = wmma_f16(a0, b0, acc[0][0]);
        acc[0][1] = wmma_f16(a0, b1, acc[0][1]);
        acc[1][0] = wmma_f16(a1, b0, acc[1][0]);
        acc[1][1] = wmma_f16(a1, b1, acc[1][1]);
    }

    int l16 = lane & 15, hi = (lane >> 4) & 1;
    bool full = (nblk < 68);                                // block 68: only n=8704 valid
#pragma unroll
    for (int ti = 0; ti < 2; ++ti) {
#pragma unroll
        for (int tj = 0; tj < 2; ++tj) {
            int colg = col_base + tj * 16 + l16;            // 0..767
            int n0 = row_base + ti * 16 + hi * 8;           // 8 contiguous rows
            float bv = bias[colg];
            v8f a = acc[ti][tj];
            if (colg < 256) {                               // q: row-major, scaled
#pragma unroll
                for (int r = 0; r < 8; ++r) {
                    int n = n0 + r;
                    if (full || n < NTOT)
                        qh[((size_t)b * NPAD + n) * CDIM + colg] =
                            (_Float16)((a[r] + bv) * QSCALE);
                }
            } else {                                        // k/v: transposed store
                _Float16* dst = (colg < 512) ? kT : vT;
                int drow = colg - ((colg < 512) ? 256 : 512);
                _Float16* p = dst + ((size_t)b * CDIM + drow) * NPAD;
                if (full) {
                    h8v hv;
#pragma unroll
                    for (int r = 0; r < 8; ++r) hv[r] = (_Float16)(a[r] + bv);
                    *(h8v*)(p + n0) = hv;                   // one b128 per lane
                } else {
#pragma unroll
                    for (int r = 0; r < 8; ++r) {
                        int n = n0 + r;
                        if (n < NTOT) p[n] = (_Float16)(a[r] + bv);
                    }
                }
            }
        }
    }
}

// ---------------- Kernel B: kv[b,h] = K^T V (split-K + f32 atomics) -------
// grid (18 chunks of 512, 4 heads, 8 batch), 256 threads = 8 waves.
// Wave computes one 16-row x 32-col slab of the 64x64 kv tile.
__global__ void __launch_bounds__(256)
kv_acc_kernel(const _Float16* __restrict__ kT,
              const _Float16* __restrict__ vT,
              float* __restrict__ kv) {
    int chunk = blockIdx.x, h = blockIdx.y, b = blockIdx.z;
    int nc = chunk * 512;
    int tid = threadIdx.x, wave = tid >> 5, lane = tid & 31;
    int tr = wave & 3, tc = (wave >> 2) * 2;
    const _Float16* kbase = kT + ((size_t)b * CDIM + h * HD) * NPAD;  // [d][n]
    const _Float16* vbase = vT + ((size_t)b * CDIM + h * HD) * NPAD;  // [e][n]

    v8f acc[2] = {};
#pragma unroll 4
    for (int k0 = 0; k0 < 512; k0 += 32) {
        v16h a  = load_afrag(kbase, NPAD, tr * 16,       nc + k0, lane);
        v16h b0 = load_bfrag(vbase, NPAD, tc * 16,       nc + k0, lane);
        v16h b1 = load_bfrag(vbase, NPAD, (tc + 1) * 16, nc + k0, lane);
        acc[0] = wmma_f16(a, b0, acc[0]);
        acc[1] = wmma_f16(a, b1, acc[1]);
    }

    int l16 = lane & 15, hi = (lane >> 4) & 1;
    float* kvb = kv + ((size_t)b * HEADS + h) * HD * HD;
#pragma unroll
    for (int j = 0; j < 2; ++j) {
        int e  = (tc + j) * 16 + l16;
        int d0 = tr * 16 + hi * 8;
#pragma unroll
        for (int r = 0; r < 8; ++r)
            unsafeAtomicAdd(&kvb[(size_t)(d0 + r) * HD + e], acc[j][r]);
    }
}

// ---------------- Kernel C: fold W_p: MT[b][e][f] = sum kv*Wp -------------
// MT[e][h*64+d] = sum_{d2} kv[b,h,d,d2] * Wp[h*64+d2][e]   (f16 out, [e][f])
__global__ void __launch_bounds__(256)
make_mt_kernel(const float* __restrict__ kv,
               const float* __restrict__ Wp,
               _Float16* __restrict__ MT) {
    int b = blockIdx.x, t = threadIdx.x;
    int h = t >> 6, d = t & 63;
    const float* kvrow = kv + (((size_t)b * HEADS + h) * HD + d) * HD;
    for (int e = 0; e < CDIM; ++e) {
        float s = 0.f;
#pragma unroll 8
        for (int d2 = 0; d2 < HD; ++d2)
            s += kvrow[d2] * Wp[(size_t)(h * HD + d2) * CDIM + e];
        MT[((size_t)b * CDIM + e) * CDIM + t] = (_Float16)s;
    }
}

// ---------------- Kernel D: out^T = MT x q^T, stored as out[b][n][e] ------
// grid (69 nblk, 4 eblk, 8 batch), 256 threads = 8 waves.
// Workgroup tile: 64 e x 128 n; wave tile 32x32.
__global__ void __launch_bounds__(256)
out_gemm_kernel(const _Float16* __restrict__ MT,
                const _Float16* __restrict__ qh,
                const float* __restrict__ bp,
                float* __restrict__ out) {
    int nblk = blockIdx.x, eblk = blockIdx.y, b = blockIdx.z;
    int tid = threadIdx.x, wave = tid >> 5, lane = tid & 31;
    int we = wave & 1, wn = wave >> 1;
    int e_base = eblk * 64 + we * 32;
    int n_base = nblk * 128 + wn * 32;
    const _Float16* Mb = MT + (size_t)b * CDIM * CDIM;      // [e][f]
    const _Float16* qb = qh + (size_t)b * NPAD * CDIM;      // [n][f]

    v8f acc[2][2] = {};
#pragma unroll
    for (int k0 = 0; k0 < CDIM; k0 += 32) {
        v16h a0 = load_afrag(Mb, CDIM, e_base,      k0, lane);
        v16h a1 = load_afrag(Mb, CDIM, e_base + 16, k0, lane);
        v16h b0 = load_bfrag(qb, CDIM, n_base,      k0, lane);
        v16h b1 = load_bfrag(qb, CDIM, n_base + 16, k0, lane);
        acc[0][0] = wmma_f16(a0, b0, acc[0][0]);
        acc[0][1] = wmma_f16(a0, b1, acc[0][1]);
        acc[1][0] = wmma_f16(a1, b0, acc[1][0]);
        acc[1][1] = wmma_f16(a1, b1, acc[1][1]);
    }

    int l16 = lane & 15, hi = (lane >> 4) & 1;
#pragma unroll
    for (int ti = 0; ti < 2; ++ti) {
#pragma unroll
        for (int tj = 0; tj < 2; ++tj) {
            int n = n_base + tj * 16 + l16;
            if (n >= NTOT) continue;
            int e0 = e_base + ti * 16 + hi * 8;             // 8 contiguous e
            const float4* bp4 = (const float4*)(bp + e0);
            float4 blo = bp4[0], bhi = bp4[1];
            v8f a = acc[ti][tj];
            float4 o0, o1;
            o0.x = a[0] + blo.x; o0.y = a[1] + blo.y;
            o0.z = a[2] + blo.z; o0.w = a[3] + blo.w;
            o1.x = a[4] + bhi.x; o1.y = a[5] + bhi.y;
            o1.z = a[6] + bhi.z; o1.w = a[7] + bhi.w;
            float* op = out + ((size_t)b * NTOT + n) * CDIM + e0;
            *(float4*)op       = o0;
            *(float4*)(op + 4) = o1;
        }
    }
}

// ---------------- launch ----------------
extern "C" void kernel_launch(void* const* d_in, const int* in_sizes, int n_in,
                              void* d_out, int out_size, void* d_ws, size_t ws_size,
                              hipStream_t stream) {
    const float* loc  = (const float*)d_in[0];
    const float* reg  = (const float*)d_in[1];
    const float* glob = (const float*)d_in[2];
    const float* Wl   = (const float*)d_in[3];
    const float* bl   = (const float*)d_in[4];
    const float* Wr   = (const float*)d_in[5];
    const float* br   = (const float*)d_in[6];
    const float* Wg   = (const float*)d_in[7];
    const float* bg   = (const float*)d_in[8];
    const float* Wp   = (const float*)d_in[9];
    const float* bp   = (const float*)d_in[10];
    float* out = (float*)d_out;

    // workspace carve-up (bytes, 256B aligned)
    char* w = (char*)d_ws;
    const size_t XH_B = (size_t)BATCH * NPAD * CDIM * 2;        // 37.75 MB
    const size_t WT_B = (size_t)3 * C3 * CDIM * 2;              // 1.18 MB
    const size_t KT_B = (size_t)BATCH * CDIM * NPAD * 2;        // 37.75 MB
    const size_t KV_B = (size_t)BATCH * HEADS * HD * HD * 4;    // 0.5 MB
    const size_t MT_B = (size_t)BATCH * CDIM * CDIM * 2;        // 1 MB
    _Float16* Xh  = (_Float16*)w;               w += (XH_B + 255) & ~255ull;
    _Float16* WhT = (_Float16*)w;               w += (WT_B + 255) & ~255ull;
    _Float16* qh  = (_Float16*)w;               w += (XH_B + 255) & ~255ull;
    _Float16* kT  = (_Float16*)w;               w += (KT_B + 255) & ~255ull;
    _Float16* vT  = (_Float16*)w;               w += (KT_B + 255) & ~255ull;
    float*    kv  = (float*)w;                  w += (KV_B + 255) & ~255ull;
    _Float16* MT  = (_Float16*)w;               w += (MT_B + 255) & ~255ull;

    // prep
    {
        size_t total4 = (size_t)BATCH * NTOT * CDIM / 4;
        conv_tokens_kernel<<<(unsigned)((total4 + 255) / 256), 256, 0, stream>>>(
            loc, reg, glob, Xh);
    }
    conv_w_kernel<<<(3 * C3 * CDIM + 255) / 256, 256, 0, stream>>>(Wl, Wr, Wg, WhT);
    {
        size_t total8 = (size_t)BATCH * CDIM * (NPAD - 8704) / 8;
        zero_tail_kernel<<<(unsigned)((total8 + 255) / 256), 256, 0, stream>>>(kT, vT);
    }
    zero_kv_kernel<<<(BATCH * HEADS * HD * HD / 4 + 255) / 256, 256, 0, stream>>>(kv);

    // 1) QKV projection  (128x64 tiles, 8 waves)
    qkv_gemm_kernel<<<dim3(C3 / 64, 69, BATCH), 256, 0, stream>>>(
        Xh, WhT, bl, br, bg, qh, kT, vT);

    // 2) kv = K^T V  (split-K over 18 chunks of 512, atomic f32 accumulate)
    kv_acc_kernel<<<dim3(NPAD / 512, HEADS, BATCH), 256, 0, stream>>>(kT, vT, kv);

    // 3) MT = (kv @ Wp)^T  per batch
    make_mt_kernel<<<BATCH, 256, 0, stream>>>(kv, Wp, MT);

    // 4) out = q @ M + bp  (computed as out^T tiles, stored row-major)
    out_gemm_kernel<<<dim3(69, CDIM / 64, BATCH), 256, 0, stream>>>(MT, qh, bp, out);
}
